// PhysicsGNNLayer_57071525429998
// MI455X (gfx1250) — compile-verified, matrix operand order checked by reading, not run on hardware
//
#include <hip/hip_runtime.h>

// ---------------------------------------------------------------------------
// Transformer encoder layer for MI455X (gfx1250, wave32, WMMA).
// All GEMMs use v_wmma_f32_16x16x32_f16 (f16 in, fp32 accumulate).
// GEMM wave tile: 32x64 (2x4 WMMA), block tile 128x128 (8 waves).
// ---------------------------------------------------------------------------

typedef _Float16 half_t;
typedef __attribute__((ext_vector_type(16))) _Float16 v16h;
typedef __attribute__((ext_vector_type(8)))  float    v8f;

#define DMODEL 512
#define NSEQ   1024
#define NBATCH 16
#define NHEAD  8
#define DHEAD  64
#define MROWS  (NBATCH * NSEQ)   // 16384 token rows

union AFrag16 { v16h v; float4 q[2]; };

// Load one 16x32 f16 WMMA fragment (A layout, or B layout for W stored [N,K]).
// Per ISA 7.12.2: lane L (0..15) holds row/col L; lanes 16..31 hold the K+8
// group. frag elems 0..7 = K kb..kb+7, elems 8..15 = K kb+16..kb+23.
__device__ __forceinline__ v16h load_frag16(const half_t* __restrict__ base,
                                            int stride, int idx, int k, int lane) {
  const int kb = (lane >> 4) << 3;   // 0 or 8
  const half_t* p = base + (size_t)idx * (size_t)stride + (size_t)(k + kb);
  AFrag16 f;
  f.q[0] = *reinterpret_cast<const float4*>(p);        // K kb .. kb+7
  f.q[1] = *reinterpret_cast<const float4*>(p + 16);   // K kb+16 .. kb+23
  return f.v;
}

__device__ __forceinline__ v8f wmma32(v16h a, v16h b, v8f c) {
  return __builtin_amdgcn_wmma_f32_16x16x32_f16(false, a, false, b,
                                                (short)0, c, false, false);
}

__device__ __forceinline__ v8f vzero8() {
  v8f z = {0.f, 0.f, 0.f, 0.f, 0.f, 0.f, 0.f, 0.f};
  return z;
}

// ---------------------------------------------------------------------------
// fp32 -> f16 conversion (weights)
// ---------------------------------------------------------------------------
__global__ void f2h_kernel(const float* __restrict__ src, half_t* __restrict__ dst, int n) {
  int i = blockIdx.x * blockDim.x + threadIdx.x;
  if (i < n) dst[i] = (half_t)src[i];
}

// ---------------------------------------------------------------------------
// LayerNorm: one wave (32 lanes) per row of 512; lane owns 16 contiguous cols.
// ---------------------------------------------------------------------------
__global__ __launch_bounds__(256) void ln_kernel(const float* __restrict__ x,
                                                 const float* __restrict__ g,
                                                 const float* __restrict__ bta,
                                                 half_t* __restrict__ out) {
  const int row  = blockIdx.x * 8 + (threadIdx.x >> 5);
  const int lane = threadIdx.x & 31;
  const float* xr = x + (size_t)row * DMODEL + lane * 16;

  float v[16];
  const float4* x4 = reinterpret_cast<const float4*>(xr);
#pragma unroll
  for (int j = 0; j < 4; ++j) {
    float4 t = x4[j];
    v[4 * j + 0] = t.x; v[4 * j + 1] = t.y; v[4 * j + 2] = t.z; v[4 * j + 3] = t.w;
  }
  float s = 0.f;
#pragma unroll
  for (int j = 0; j < 16; ++j) s += v[j];
#pragma unroll
  for (int o = 16; o > 0; o >>= 1) s += __shfl_xor(s, o);
  const float mu = s * (1.0f / DMODEL);

  float q = 0.f;
#pragma unroll
  for (int j = 0; j < 16; ++j) { float d = v[j] - mu; q += d * d; }
#pragma unroll
  for (int o = 16; o > 0; o >>= 1) q += __shfl_xor(q, o);
  const float rs = rsqrtf(q * (1.0f / DMODEL) + 1e-5f);

  half_t* orow = out + (size_t)row * DMODEL + lane * 16;
#pragma unroll
  for (int j = 0; j < 16; ++j) {
    int c = lane * 16 + j;
    orow[j] = (half_t)((v[j] - mu) * rs * g[c] + bta[c]);
  }
}

// ---------------------------------------------------------------------------
// WMMA GEMM: out[M,Nn] = act(A_f16[M,K] @ W_f16[Nn,K]^T + bias) (+ resid)
// Block: 256 thr = 8 waves; block tile 128x128; wave tile 32x64 (2x4 WMMA).
// 8 WMMAs per 6 fragment loads -> 1.33x register-level reuse vs 2x2 tiling.
// ---------------------------------------------------------------------------
__global__ __launch_bounds__(256) void gemm_kernel(const half_t* __restrict__ A,
                                                   const half_t* __restrict__ W,
                                                   const float* __restrict__ bias,
                                                   const float* __restrict__ resid,
                                                   float* __restrict__ out32,
                                                   half_t* __restrict__ out16,
                                                   int Nn, int K, int act) {
  const int lane = threadIdx.x & 31;
  const int w    = threadIdx.x >> 5;
  const int wm   = w & 3;          // 4 waves along M (32 rows each)
  const int wn   = w >> 2;         // 2 waves along N (64 cols each)
  const int m0   = blockIdx.y * 128 + wm * 32;
  const int n0   = blockIdx.x * 128 + wn * 64;
  const int lm   = lane & 15;
  const int hi   = lane >> 4;

  v8f acc[2][4];
#pragma unroll
  for (int i = 0; i < 2; ++i)
#pragma unroll
    for (int j = 0; j < 4; ++j) acc[i][j] = vzero8();

  for (int k = 0; k < K; k += 32) {
    v16h a0 = load_frag16(A, K, m0 + lm,      k, lane);
    v16h a1 = load_frag16(A, K, m0 + 16 + lm, k, lane);
    v16h b0 = load_frag16(W, K, n0 + lm,      k, lane);
    v16h b1 = load_frag16(W, K, n0 + 16 + lm, k, lane);
    v16h b2 = load_frag16(W, K, n0 + 32 + lm, k, lane);
    v16h b3 = load_frag16(W, K, n0 + 48 + lm, k, lane);
    acc[0][0] = wmma32(a0, b0, acc[0][0]);
    acc[0][1] = wmma32(a0, b1, acc[0][1]);
    acc[0][2] = wmma32(a0, b2, acc[0][2]);
    acc[0][3] = wmma32(a0, b3, acc[0][3]);
    acc[1][0] = wmma32(a1, b0, acc[1][0]);
    acc[1][1] = wmma32(a1, b1, acc[1][1]);
    acc[1][2] = wmma32(a1, b2, acc[1][2]);
    acc[1][3] = wmma32(a1, b3, acc[1][3]);
  }

#pragma unroll
  for (int i = 0; i < 2; ++i) {
#pragma unroll
    for (int j = 0; j < 4; ++j) {
      const int colg = n0 + j * 16 + lm;
      const float bz = bias[colg];
#pragma unroll
      for (int r = 0; r < 8; ++r) {
        const int rowg = m0 + i * 16 + r + (hi << 3);
        float val = acc[i][j][r] + bz;
        if (act) val = val / (1.f + __expf(-val));     // SiLU
        const size_t off = (size_t)rowg * Nn + colg;
        if (resid) val += resid[off];
        if (out32) out32[off] = val;
        if (out16) out16[off] = (half_t)val;
      }
    }
  }
}

// ---------------------------------------------------------------------------
// V [B,N,H,DH] -> Vt [B,H,DH,N]
// ---------------------------------------------------------------------------
__global__ void transpose_v_kernel(const half_t* __restrict__ v, half_t* __restrict__ vt) {
  size_t i = (size_t)blockIdx.x * 256 + threadIdx.x;   // indexes vt, n fastest
  int n  = (int)(i & (NSEQ - 1));
  size_t r = i >> 10;
  int d  = (int)(r & (DHEAD - 1)); r >>= 6;
  int hh = (int)(r & (NHEAD - 1));
  int b  = (int)(r >> 3);
  vt[i] = v[(((size_t)(b * NSEQ + n)) * NHEAD + hh) * DHEAD + d];
}

// ---------------------------------------------------------------------------
// Attention: one wave per (b, h, 16-query tile).
// LDS: S = 16x1024 f32 scores (64KB), P = 16x1024 f16 probs (32KB) -> 96KB.
// ---------------------------------------------------------------------------
__global__ __launch_bounds__(32) void attn_kernel(const half_t* __restrict__ q,
                                                  const half_t* __restrict__ kk,
                                                  const half_t* __restrict__ vt,
                                                  half_t* __restrict__ ctx,
                                                  float* __restrict__ attnw) {
  extern __shared__ char smem[];
  float*  S = (float*)smem;                                // [16][NSEQ]
  half_t* P = (half_t*)(smem + 16 * NSEQ * sizeof(float)); // [16][NSEQ]

  const int lane = threadIdx.x;
  const int bid  = blockIdx.x;
  const int qt = bid & 63;          // N/16 = 64 query tiles
  const int hh = (bid >> 6) & 7;
  const int b  = bid >> 9;
  const int q0 = qt * 16;
  const int m  = lane & 15;
  const int hi = lane >> 4;

  // Q fragments for this 16-row query tile (K = DH = 64 -> 2 frags)
  const half_t* qbase = q + (((size_t)(b * NSEQ + q0 + m)) * NHEAD + hh) * DHEAD;
  v16h aq0 = load_frag16(qbase, 0, 0, 0,  lane);
  v16h aq1 = load_frag16(qbase, 0, 0, 32, lane);

  // ---- scores: S[16, NSEQ] = (Q @ K^T) / sqrt(DH) ----
  for (int nt = 0; nt < NSEQ / 16; ++nt) {
    const int n0 = nt * 16;
    const half_t* kbase = kk + (((size_t)(b * NSEQ + n0 + m)) * NHEAD + hh) * DHEAD;
    v16h bk0 = load_frag16(kbase, 0, 0, 0,  lane);
    v16h bk1 = load_frag16(kbase, 0, 0, 32, lane);
    v8f sacc = vzero8();
    sacc = wmma32(aq0, bk0, sacc);
    sacc = wmma32(aq1, bk1, sacc);
#pragma unroll
    for (int r = 0; r < 8; ++r)
      S[(r + (hi << 3)) * NSEQ + n0 + m] = sacc[r] * 0.125f;   // 1/sqrt(64)
  }
  __syncthreads();

  // ---- softmax: lane pair (l, l+16) shares row m, split columns ----
  {
    const int row = m;
    const int c0  = hi * (NSEQ / 2);
    float* Srow = S + row * NSEQ + c0;
    float mx = -1e30f;
    for (int c = 0; c < NSEQ / 2; ++c) mx = fmaxf(mx, Srow[c]);
    mx = fmaxf(mx, __shfl_xor(mx, 16));
    float sum = 0.f;
    for (int c = 0; c < NSEQ / 2; ++c) {
      float e = __expf(Srow[c] - mx);
      Srow[c] = e;
      sum += e;
    }
    sum += __shfl_xor(sum, 16);
    const float inv = 1.f / sum;
    const float wsc = inv * (1.0f / NHEAD);   // head-average factor
    float* aw = attnw + ((size_t)(b * NSEQ + q0 + row)) * NSEQ + c0;
    half_t* Prow = P + row * NSEQ + c0;
    for (int c = 0; c < NSEQ / 2; ++c) {
      float e = Srow[c];
      Prow[c] = (half_t)(e * inv);
      atomicAdd(&aw[c], e * wsc);
    }
  }
  __syncthreads();

  // ---- ctx: P[16, NSEQ] @ V[NSEQ, DH] using Vt rows (contiguous keys) ----
  v8f cacc[4];
#pragma unroll
  for (int d = 0; d < 4; ++d) cacc[d] = vzero8();

  const half_t* vbase = vt + ((size_t)(b * NHEAD + hh)) * DHEAD * NSEQ;
  for (int k0 = 0; k0 < NSEQ; k0 += 32) {
    v16h ap = load_frag16(P, NSEQ, m, k0, lane);
#pragma unroll
    for (int d = 0; d < 4; ++d) {
      v16h bv = load_frag16(vbase, NSEQ, d * 16 + m, k0, lane);
      cacc[d] = wmma32(ap, bv, cacc[d]);
    }
  }
#pragma unroll
  for (int d = 0; d < 4; ++d) {
#pragma unroll
    for (int r = 0; r < 8; ++r) {
      const int rq = q0 + r + (hi << 3);
      ctx[(((size_t)(b * NSEQ + rq)) * NHEAD + hh) * DHEAD + d * 16 + m] = (half_t)cacc[d][r];
    }
  }
}

// ---------------------------------------------------------------------------
// Host launch
// ---------------------------------------------------------------------------
extern "C" void kernel_launch(void* const* d_in, const int* in_sizes, int n_in,
                              void* d_out, int out_size, void* d_ws, size_t ws_size,
                              hipStream_t stream) {
  (void)in_sizes; (void)n_in; (void)out_size; (void)ws_size;

  const float* h_in  = (const float*)d_in[0];
  const float* ln1_g = (const float*)d_in[1];
  const float* ln1_b = (const float*)d_in[2];
  const float* wq = (const float*)d_in[3];  const float* bq = (const float*)d_in[4];
  const float* wk = (const float*)d_in[5];  const float* bk = (const float*)d_in[6];
  const float* wv = (const float*)d_in[7];  const float* bv = (const float*)d_in[8];
  const float* wo = (const float*)d_in[9];  const float* bo = (const float*)d_in[10];
  const float* ln2_g = (const float*)d_in[11]; const float* ln2_b = (const float*)d_in[12];
  const float* w1 = (const float*)d_in[13]; const float* b1 = (const float*)d_in[14];
  const float* w2 = (const float*)d_in[15]; const float* b2 = (const float*)d_in[16];

  // ---- workspace layout (bytes) ----
  char* ws = (char*)d_ws;
  size_t off = 0;
  half_t* WQH = (half_t*)(ws + off); off += (size_t)DMODEL * DMODEL * 2;        // 0.5 MB
  half_t* WKH = (half_t*)(ws + off); off += (size_t)DMODEL * DMODEL * 2;
  half_t* WVH = (half_t*)(ws + off); off += (size_t)DMODEL * DMODEL * 2;
  half_t* WOH = (half_t*)(ws + off); off += (size_t)DMODEL * DMODEL * 2;
  half_t* W1H = (half_t*)(ws + off); off += (size_t)2 * DMODEL * DMODEL * 2;    // (2D,D)
  half_t* W2H = (half_t*)(ws + off); off += (size_t)2 * DMODEL * DMODEL * 2;    // (D,2D)
  half_t* HN  = (half_t*)(ws + off); off += (size_t)MROWS * DMODEL * 2;         // hn / f (reused)
  float*  H2  = (float*)(ws + off);  off += (size_t)MROWS * DMODEL * 4;         // h + attn_out
  half_t* CTX = (half_t*)(ws + off); off += (size_t)MROWS * DMODEL * 2;
  half_t* Qb  = (half_t*)(ws + off); size_t qoff = off; off += (size_t)MROWS * DMODEL * 2;
  half_t* Kb  = (half_t*)(ws + off); off += (size_t)MROWS * DMODEL * 2;
  half_t* Vb  = (half_t*)(ws + off); off += (size_t)MROWS * DMODEL * 2;
  half_t* VT  = (half_t*)(ws + off); off += (size_t)MROWS * DMODEL * 2;
  half_t* Gb  = (half_t*)(ws + qoff);  // FFN hidden [M, 2D] reuses Q+K regions (dead by then)

  float* out_h    = (float*)d_out;
  float* out_attn = (float*)d_out + (size_t)MROWS * DMODEL;

  // attn_weights accumulated via atomics -> zero it first (capture-safe).
  hipMemsetAsync(out_attn, 0, (size_t)NBATCH * NSEQ * NSEQ * sizeof(float), stream);

  // weights -> f16
  const int DD = DMODEL * DMODEL, DD2 = 2 * DMODEL * DMODEL;
  f2h_kernel<<<(DD  + 255) / 256, 256, 0, stream>>>(wq, WQH, DD);
  f2h_kernel<<<(DD  + 255) / 256, 256, 0, stream>>>(wk, WKH, DD);
  f2h_kernel<<<(DD  + 255) / 256, 256, 0, stream>>>(wv, WVH, DD);
  f2h_kernel<<<(DD  + 255) / 256, 256, 0, stream>>>(wo, WOH, DD);
  f2h_kernel<<<(DD2 + 255) / 256, 256, 0, stream>>>(w1, W1H, DD2);
  f2h_kernel<<<(DD2 + 255) / 256, 256, 0, stream>>>(w2, W2H, DD2);

  // LN1
  ln_kernel<<<MROWS / 8, 256, 0, stream>>>(h_in, ln1_g, ln1_b, HN);

  // QKV projections (block tile 128x128)
  dim3 gD(DMODEL / 128, MROWS / 128);       // (4, 128)
  gemm_kernel<<<gD, 256, 0, stream>>>(HN, WQH, bq, nullptr, nullptr, Qb, DMODEL, DMODEL, 0);
  gemm_kernel<<<gD, 256, 0, stream>>>(HN, WKH, bk, nullptr, nullptr, Kb, DMODEL, DMODEL, 0);
  gemm_kernel<<<gD, 256, 0, stream>>>(HN, WVH, bv, nullptr, nullptr, Vb, DMODEL, DMODEL, 0);

  // V -> Vt
  transpose_v_kernel<<<(MROWS * DMODEL) / 256, 256, 0, stream>>>(Vb, VT);

  // attention (96 KB dynamic LDS; CDNA5 WGP has 320 KB)
  const int attn_lds = 16 * NSEQ * (int)sizeof(float) + 16 * NSEQ * (int)sizeof(half_t);
  hipFuncSetAttribute(reinterpret_cast<const void*>(attn_kernel),
                      hipFuncAttributeMaxDynamicSharedMemorySize, attn_lds);
  attn_kernel<<<NBATCH * NHEAD * (NSEQ / 16), 32, attn_lds, stream>>>(Qb, Kb, VT, CTX, out_attn);

  // out projection + residual -> h2 (fp32)
  gemm_kernel<<<gD, 256, 0, stream>>>(CTX, WOH, bo, h_in, H2, nullptr, DMODEL, DMODEL, 0);

  // LN2 -> f (f16, reuses HN)
  ln_kernel<<<MROWS / 8, 256, 0, stream>>>(H2, ln2_g, ln2_b, HN);

  // FFN1: silu(f @ w1^T + b1) -> g f16 [M, 2D]
  dim3 g2(2 * DMODEL / 128, MROWS / 128);   // (8, 128)
  gemm_kernel<<<g2, 256, 0, stream>>>(HN, W1H, b1, nullptr, nullptr, Gb, 2 * DMODEL, DMODEL, 1);

  // FFN2: g @ w2^T + b2 + h2 -> final h (fp32, straight to d_out)
  gemm_kernel<<<gD, 256, 0, stream>>>(Gb, W2H, b2, H2, out_h, nullptr, DMODEL, 2 * DMODEL, 0);
}
